// SingleHeadAttention_5557687681479
// MI455X (gfx1250) — matrix operand order repaired and checked by reading, compiled, not run on hardware
//
#include <hip/hip_runtime.h>

// Problem dims (fixed by reference)
#define BATCH 256
#define TDIM  256
#define CDIM  384
#define HDIM  64
#define WAVES1 8        // 256 threads, 128 rows per projection block
#define WAVES2 4        // 128 threads per attention block
#define BQ    64        // q rows per attention block

typedef __attribute__((ext_vector_type(16))) __bf16 v16bf;
typedef __attribute__((ext_vector_type(8)))  float  v8f;

union AB {
  v16bf v;
  unsigned short h[16];
  unsigned int   w[8];
  uint4          q[2];
};

__device__ __forceinline__ unsigned short f2bf(float f) {
  __bf16 b = (__bf16)f;
  return __builtin_bit_cast(unsigned short, b);
}
__device__ __forceinline__ unsigned int pack2(float a, float b) {
  return (unsigned int)f2bf(a) | ((unsigned int)f2bf(b) << 16);
}
__device__ __forceinline__ void ld2x128(AB& dst, const uint4* p, int base) {
  dst.q[0] = p[base];
  dst.q[1] = p[base + 1];
}

// ---------------------------------------------------------------------------
// Kernel 1: q/k/v = x @ {Wq,Wk,Wv}, bf16 output to workspace.
// Weights staged transposed in LDS as bf16: Wt[m][col][k]; B-fragments are
// two contiguous ds_load_b128, pipelined at distance 2 over 3 buffers so
// each WMMA overlaps two fragments' worth of LDS latency.
// ---------------------------------------------------------------------------
__global__ void qkv_proj_kernel(const float* __restrict__ x,
                                const float* __restrict__ Wq,
                                const float* __restrict__ Wk,
                                const float* __restrict__ Wv,
                                unsigned short* __restrict__ qkv) {
  extern __shared__ unsigned short Wt[];  // [3][64][384] ush = 144 KB

  const int tid  = threadIdx.x;
  const int lane = tid & 31;
  const int wave = tid >> 5;
  const int lm   = lane & 15;      // lane-in-half
  const int lh   = lane >> 4;      // lane half (0/1)
  const int m0   = blockIdx.x * (WAVES1 * 16) + wave * 16;

  // ---- stage weights (f32 global, coalesced b128) -> bf16 transposed LDS ---
  for (int i = tid; i < 18432; i += WAVES1 * 32) {   // 3*384*16 float4
    const int m   = i / 6144;
    const int rem = i - m * 6144;
    const int k   = rem >> 4;          // 0..383
    const int c4  = rem & 15;          // float4 along col
    const float* wsrc = (m == 0) ? Wq : ((m == 1) ? Wk : Wv);
    const float4 d = ((const float4*)wsrc)[k * 16 + c4];
    unsigned short* dst = Wt + m * 24576 + (c4 * 4) * CDIM + k;
    dst[0 * CDIM] = f2bf(d.x);
    dst[1 * CDIM] = f2bf(d.y);
    dst[2 * CDIM] = f2bf(d.z);
    dst[3 * CDIM] = f2bf(d.w);
  }
  __syncthreads();

  v8f zero = {};
  v8f acc[3][4];
#pragma unroll
  for (int m = 0; m < 3; ++m)
#pragma unroll
    for (int n = 0; n < 4; ++n) acc[m][n] = zero;

  const int arow = m0 + lm;
  const float4* x4 = (const float4*)(x + (size_t)arow * CDIM);
  const uint4* wt4 = (const uint4*)Wt;

#pragma unroll 1
  for (int kc = 0; kc < CDIM / 32; ++kc) {
    if (kc + 1 < CDIM / 32)
      __builtin_prefetch((const float*)x4 + (kc + 1) * 32, 0, 3);

    // A fragment: lane's K range is one contiguous 16-float run.
    const int xb = kc * 8 + lh * 4;       // float4 index
    const float4 f0 = x4[xb];
    const float4 f1 = x4[xb + 1];
    const float4 f2 = x4[xb + 2];
    const float4 f3 = x4[xb + 3];
    AB a;
    a.w[0] = pack2(f0.x, f0.y);
    a.w[1] = pack2(f0.z, f0.w);
    a.w[2] = pack2(f1.x, f1.y);
    a.w[3] = pack2(f1.z, f1.w);
    a.w[4] = pack2(f2.x, f2.y);
    a.w[5] = pack2(f2.z, f2.w);
    a.w[6] = pack2(f3.x, f3.y);
    a.w[7] = pack2(f3.z, f3.w);

    // 12 B fragments (m*4+nt), distance-2 pipelined over 3 buffers.
    const int kb0 = kc * 32 + lh * 16;
    AB bb[3];
    ld2x128(bb[0], wt4, (0 * 24576 + (0 * 16 + lm) * CDIM + kb0) >> 3);
    ld2x128(bb[1], wt4, (0 * 24576 + (1 * 16 + lm) * CDIM + kb0) >> 3);
#pragma unroll
    for (int f = 0; f < 12; ++f) {
      if (f + 2 < 12) {
        const int m2   = (f + 2) >> 2;
        const int nt2  = (f + 2) & 3;
        ld2x128(bb[(f + 2) % 3], wt4,
                (m2 * 24576 + (nt2 * 16 + lm) * CDIM + kb0) >> 3);
      }
      acc[f >> 2][f & 3] = __builtin_amdgcn_wmma_f32_16x16x32_bf16(
          false, a.v, false, bb[f % 3].v, (short)0, acc[f >> 2][f & 3], false,
          false);
    }
  }

  // Store bf16: D layout row = r + 8*laneHalf, col = nt*16 + lm.
#pragma unroll
  for (int m = 0; m < 3; ++m) {
    unsigned short* dst = qkv + (size_t)m * (BATCH * TDIM * HDIM);
#pragma unroll
    for (int nt = 0; nt < 4; ++nt) {
      const int col = nt * 16 + lm;
#pragma unroll
      for (int r = 0; r < 8; ++r) {
        const int row = m0 + r + lh * 8;
        dst[(size_t)row * HDIM + col] = f2bf(acc[m][nt][r]);
      }
    }
  }
}

// ---------------------------------------------------------------------------
// Kernel 2: causal attention for one (batch, 64-row q tile).
// LDS: Ks[256][64] bf16, Vt[64][256] bf16 (transposed), Ps[4][16][256] bf16.
// K is staged with async global->LDS (ASYNCcnt); no static LDS in this
// kernel, so the dynamic-LDS byte offset is the LDS address directly.
// ---------------------------------------------------------------------------
__global__ void attn_kernel(const unsigned short* __restrict__ qkv,
                            float* __restrict__ out, float scale) {
  extern __shared__ unsigned short smem[];
  unsigned short* Ks = smem;                 // 16384 ush, LDS byte off 0
  unsigned short* Vt = smem + 16384;         // 16384 ush
  unsigned short* Ps = smem + 32768;         // 4*4096 ush

  const int tid  = threadIdx.x;
  const int lane = tid & 31;
  const int wave = tid >> 5;
  const int lm   = lane & 15;
  const int lh   = lane >> 4;
  const int b    = blockIdx.y;
  const int q0   = blockIdx.x * BQ + wave * 16;

  const unsigned short* qg = qkv;
  const unsigned short* kg = qkv + (size_t)(BATCH * TDIM * HDIM);
  const unsigned short* vg = qkv + (size_t)2 * (BATCH * TDIM * HDIM);

  // Async K stage (row major): 2048 x b128, memory -> LDS without VGPRs.
  {
    const uint4* ksrc = (const uint4*)(kg + (size_t)b * TDIM * HDIM);
    for (int i = tid; i < 2048; i += blockDim.x) {
      const unsigned int lds_off = (unsigned int)i * 16u;  // Ks starts at 0
      const unsigned long long gaddr = (unsigned long long)(ksrc + i);
      asm volatile("global_load_async_to_lds_b128 %0, %1, off"
                   :
                   : "v"(lds_off), "v"(gaddr)
                   : "memory");
    }
  }
  // Cooperative V stage, transposed: Vt[h][key].
  {
    const uint4* vsrc = (const uint4*)(vg + (size_t)b * TDIM * HDIM);
    for (int i = tid; i < 2048; i += blockDim.x) {
      uint4 d = vsrc[i];
      const int key = i >> 3;          // 8 uint4 per 64-wide row
      const int h0  = (i & 7) * 8;
      const unsigned short* e = (const unsigned short*)&d;
#pragma unroll
      for (int j = 0; j < 8; ++j) Vt[(h0 + j) * TDIM + key] = e[j];
    }
  }
  asm volatile("s_wait_asynccnt 0x0" ::: "memory");
  __syncthreads();

  v8f zero = {};

  // Q fragments for both 32-wide H chunks (pairs are contiguous -> dword loads)
  AB aq[2];
  {
    const unsigned int* qw =
        (const unsigned int*)(qg + ((size_t)b * TDIM + q0 + lm) * HDIM);
#pragma unroll
    for (int c = 0; c < 2; ++c)
#pragma unroll
      for (int p = 0; p < 8; ++p) {
        const int kb = c * 32 + lh * 16 + p * 2;
        aq[c].w[p] = qw[kb >> 1];
      }
  }

  // Scores: S = Q K^T over H=64; 32 (kt,c) WMMA steps, distance-2 pipeline.
  v8f S[16];
  const uint4* ks4 = (const uint4*)Ks;
  {
    AB bb[3];
    ld2x128(bb[0], ks4, (lm * HDIM + lh * 16) >> 3);            // kt=0,c=0
    ld2x128(bb[1], ks4, (lm * HDIM + 32 + lh * 16) >> 3);       // kt=0,c=1
#pragma unroll
    for (int s = 0; s < 32; ++s) {
      const int kt = s >> 1;
      const int c  = s & 1;
      if (s + 2 < 32) {
        const int kt2 = (s + 2) >> 1;
        const int c2  = (s + 2) & 1;
        ld2x128(bb[(s + 2) % 3], ks4,
                ((kt2 * 16 + lm) * HDIM + c2 * 32 + lh * 16) >> 3);
      }
      if (c == 0) S[kt] = zero;
      S[kt] = __builtin_amdgcn_wmma_f32_16x16x32_bf16(
          false, aq[c].v, false, bb[s % 3].v, (short)0, S[kt], false, false);
    }
  }

  // Scale, causal mask, row softmax (row = r + 8*lh, cols across 16 lanes).
#pragma unroll
  for (int r = 0; r < 8; ++r) {
    const int qrow = q0 + r + lh * 8;
    float mx = -INFINITY;
#pragma unroll
    for (int kt = 0; kt < 16; ++kt) {
      const int col = kt * 16 + lm;
      float s = (col <= qrow) ? S[kt][r] * scale : -INFINITY;
      S[kt][r] = s;
      mx = fmaxf(mx, s);
    }
#pragma unroll
    for (int off = 8; off >= 1; off >>= 1)
      mx = fmaxf(mx, __shfl_xor(mx, off, 32));
    float sum = 0.f;
#pragma unroll
    for (int kt = 0; kt < 16; ++kt) {
      const float p = __expf(S[kt][r] - mx);
      S[kt][r] = p;
      sum += p;
    }
#pragma unroll
    for (int off = 8; off >= 1; off >>= 1) sum += __shfl_xor(sum, off, 32);
    const float inv = 1.0f / sum;
    const int prow = r + lh * 8;
    unsigned short* pr = Ps + wave * 4096 + prow * 256;
#pragma unroll
    for (int kt = 0; kt < 16; ++kt) pr[kt * 16 + lm] = f2bf(S[kt][r] * inv);
  }
  __syncthreads();

  // Out = P @ V over 256 keys; 32 (nt,kc) WMMA steps, distance-2 pipeline.
  const uint4* vt4 = (const uint4*)Vt;
  const unsigned int* pw = (const unsigned int*)(Ps + wave * 4096 + lm * 256);
  {
    AB pa[3], vb[3];
#pragma unroll
    for (int pre = 0; pre < 2; ++pre) {   // preload s=0 (nt0,kc0), s=1 (nt0,kc1)
#pragma unroll
      for (int p = 0; p < 8; ++p) {
        const int kb = pre * 32 + lh * 16 + p * 2;
        pa[pre].w[p] = pw[kb >> 1];
      }
      ld2x128(vb[pre], vt4, (lm * TDIM + pre * 32 + lh * 16) >> 3);
    }
    v8f oacc = zero;
#pragma unroll
    for (int s = 0; s < 32; ++s) {
      const int nt = s >> 3;
      const int kc = s & 7;
      if (s + 2 < 32) {
        const int nt2 = (s + 2) >> 3;
        const int kc2 = (s + 2) & 7;
        const int sl  = (s + 2) % 3;
#pragma unroll
        for (int p = 0; p < 8; ++p) {
          const int kb = kc2 * 32 + lh * 16 + p * 2;
          pa[sl].w[p] = pw[kb >> 1];
        }
        ld2x128(vb[sl], vt4, ((nt2 * 16 + lm) * TDIM + kc2 * 32 + lh * 16) >> 3);
      }
      oacc = __builtin_amdgcn_wmma_f32_16x16x32_bf16(
          false, pa[s % 3].v, false, vb[s % 3].v, (short)0, oacc, false, false);
      if (kc == 7) {
        const int hcol = nt * 16 + lm;
#pragma unroll
        for (int r = 0; r < 8; ++r) {
          const int row = q0 + r + lh * 8;
          out[((size_t)b * TDIM + row) * HDIM + hcol] = oacc[r];
        }
        oacc = zero;
      }
    }
  }
}

// ---------------------------------------------------------------------------
extern "C" void kernel_launch(void* const* d_in, const int* in_sizes, int n_in,
                              void* d_out, int out_size, void* d_ws,
                              size_t ws_size, hipStream_t stream) {
  const float* x  = (const float*)d_in[0];
  const float* Wq = (const float*)d_in[1];
  const float* Wk = (const float*)d_in[2];
  const float* Wv = (const float*)d_in[3];
  float* out = (float*)d_out;
  unsigned short* qkv = (unsigned short*)d_ws;  // bf16 q|k|v, 3 * B*T*H

  dim3 g1((BATCH * TDIM) / (WAVES1 * 16));
  const size_t smem1 = (size_t)3 * 64 * CDIM * sizeof(unsigned short);  // 144 KB
  qkv_proj_kernel<<<g1, WAVES1 * 32, smem1, stream>>>(x, Wq, Wk, Wv, qkv);

  dim3 g2(TDIM / BQ, BATCH);
  const size_t smem2 =
      (size_t)(16384 + 16384 + WAVES2 * 4096) * sizeof(unsigned short);  // 96 KB
  // scale = C^-0.5 = 384^-0.5 (reference scales by n_embed, not head size)
  attn_kernel<<<g2, WAVES2 * 32, smem2, stream>>>(qkv, out,
                                                  0.05103103630798288f);
}
